// AdditiveAttention_15590731285022
// MI455X (gfx1250) — matrix operand order repaired and checked by reading, not compiled, not run on hardware
//
#include <hip/hip_runtime.h>
#include <hip/hip_bf16.h>

#define H 1024
#define B 64
#define S 2048

typedef __attribute__((ext_vector_type(16))) __bf16 v16bf;
typedef __attribute__((ext_vector_type(8)))  __bf16 bf16x8;
typedef __attribute__((ext_vector_type(8)))  float  v8f;

__device__ __forceinline__ unsigned short f2bf(float f) {
    unsigned int u = __float_as_uint(f);
    unsigned int r = (u + 0x7FFFu + ((u >> 16) & 1u)) >> 16;  // round-to-nearest-even
    return (unsigned short)r;
}

// Fast tanh: prefer the gfx1250 V_TANH trans op; otherwise branch-free
// tanh(x) = 1 - 2/(exp(2x)+1) on hardware v_exp_f32 / v_rcp_f32.
// Saturates correctly: x -> +inf => 1, x -> -inf => -1.
__device__ __forceinline__ float fast_tanh(float x) {
#if __has_builtin(__builtin_amdgcn_tanhf)
    return __builtin_amdgcn_tanhf(x);
#else
    const float e = __builtin_amdgcn_exp2f(x * 2.8853900817779268f);  // 2*log2(e)*x
    return 1.0f - 2.0f * __builtin_amdgcn_rcpf(e + 1.0f);
#endif
}

// ---------------------------------------------------------------- prep: We -> bf16
__global__ __launch_bounds__(256) void conv_we_bf16(const float* __restrict__ We,
                                                    unsigned short* __restrict__ we_bf) {
    const int i = (blockIdx.x * 256 + threadIdx.x) * 4;
    const float4 f = *(const float4*)(We + i);
    uint2 p;
    p.x = (unsigned)f2bf(f.x) | ((unsigned)f2bf(f.y) << 16);
    p.y = (unsigned)f2bf(f.z) | ((unsigned)f2bf(f.w) << 16);
    *(uint2*)(we_bf + i) = p;
}

// ---------------------------------------------------------------- prep: proj_prev = d @ Wp^T
__global__ __launch_bounds__(256) void proj_prev_kernel(const float* __restrict__ dstate,
                                                        const float* __restrict__ Wp,
                                                        float* __restrict__ out) {
    const int wave = threadIdx.x >> 5, lane = threadIdx.x & 31;
    const int o = blockIdx.x * 8 + wave;          // o = b*H + h
    const int b = o >> 10, h = o & (H - 1);
    const float* dp = dstate + (size_t)b * H;
    const float* wp = Wp + (size_t)h * H;
    float acc = 0.f;
    for (int k = lane; k < H; k += 32) acc += dp[k] * wp[k];
    #pragma unroll
    for (int off = 16; off >= 1; off >>= 1) acc += __shfl_xor(acc, off, 32);
    if (lane == 0) out[o] = acc;
}

// ---------------------------------------------------------------- main fused scores kernel
// One block per 16 rows of the flattened (s*B+b, H) row space.
__global__ __launch_bounds__(256) void scores_kernel(const float* __restrict__ enc,
                                                     const unsigned short* __restrict__ we_bf,
                                                     const float* __restrict__ pprev,
                                                     const float* __restrict__ v,
                                                     float* __restrict__ scores) {
    // bf16 row stride 1032 -> 2064B -> 516 dwords -> 4-bank rotation/row: conflict-free b128
    __shared__ unsigned short sA[16][H + 8];
    __shared__ float sPart[8][16];

    const int tid = threadIdx.x;
    const size_t rowbase = (size_t)blockIdx.x * 16;

    // Stage 16 fp32 rows -> bf16 LDS (each iteration: one float4 -> 4 bf16)
    for (int i = tid; i < 16 * (H / 4); i += 256) {
        const int row = i >> 8;                 // H/4 == 256 quads per row
        const int c4  = i & 255;
        const float4 f = *(const float4*)(enc + (rowbase + row) * H + (size_t)c4 * 4);
        uint2 p;
        p.x = (unsigned)f2bf(f.x) | ((unsigned)f2bf(f.y) << 16);
        p.y = (unsigned)f2bf(f.z) | ((unsigned)f2bf(f.w) << 16);
        *(uint2*)&sA[row][c4 * 4] = p;
    }
    __syncthreads();

    const int wave = tid >> 5, lane = tid & 31;
    const int nlo = lane & 15;                  // row (A) / column (B,D) within tile
    const int nhi = lane >> 4;                  // lane half
    const int mbase = nhi * 8;                  // D rows held by this lane half

    float sacc[8];
    #pragma unroll
    for (int r = 0; r < 8; ++r) sacc[r] = 0.f;

    for (int j = 0; j < 8; ++j) {               // this wave's 8 N-tiles
        const int nt = wave * 8 + j;
        const int n  = nt * 16 + nlo;           // output column / We row
        const unsigned short* brow = we_bf + (size_t)n * H;

        v8f acc = {0.f, 0.f, 0.f, 0.f, 0.f, 0.f, 0.f, 0.f};
        union { v16bf v; bf16x8 h[2]; } a, bm;

        #pragma unroll 8
        for (int k = 0; k < 32; ++k) {
            // A fragment (16x32 bf16) from LDS, ISA layout
            const unsigned short* ap = &sA[nlo][k * 32 + nhi * 8];
            a.h[0] = *(const bf16x8*)(const void*)(ap);
            a.h[1] = *(const bf16x8*)(const void*)(ap + 16);
            // B fragment (32x16 bf16): K 8h..8h+7 and 16+8h..16+8h+7 of one We row
            const unsigned short* bp = brow + (size_t)(k * 32 + nhi * 8);
            bm.h[0] = *(const bf16x8*)(const void*)(bp);
            bm.h[1] = *(const bf16x8*)(const void*)(bp + 16);
            acc = __builtin_amdgcn_wmma_f32_16x16x32_bf16(
                false, a.v, false, bm.v, (short)0, acc, false, false);
        }

        // fused epilogue: score += v[n] * tanh(D + proj_prev[b][n])
        const float vn = v[n];
        #pragma unroll
        for (int r = 0; r < 8; ++r) {
            const int m = mbase + r;                     // D VGPR r <-> M = r (+8 for hi half)
            const int bidx = (int)((rowbase + m) & (B - 1));  // row = s*B + b, B=64
            const float pp = pprev[(size_t)bidx * H + n];
            sacc[r] += vn * fast_tanh(acc[r] + pp);
        }
    }

    // reduce over the 16 lanes of each half (n dimension)
    #pragma unroll
    for (int r = 0; r < 8; ++r) {
        float x = sacc[r];
        x += __shfl_xor(x, 1, 32);
        x += __shfl_xor(x, 2, 32);
        x += __shfl_xor(x, 4, 32);
        x += __shfl_xor(x, 8, 32);
        sacc[r] = x;
    }
    if (nlo == 0) {
        #pragma unroll
        for (int r = 0; r < 8; ++r) sPart[wave][mbase + r] = sacc[r];
    }
    __syncthreads();
    if (tid < 16) {                              // fixed-order cross-wave sum: deterministic
        float s = 0.f;
        #pragma unroll
        for (int w = 0; w < 8; ++w) s += sPart[w][tid];
        scores[rowbase + tid] = s;
    }
}

// ---------------------------------------------------------------- masked softmax per batch
__global__ __launch_bounds__(256) void softmax_kernel(const float* __restrict__ scores,
                                                      const int* __restrict__ mask,
                                                      float* __restrict__ attn) {
    __shared__ float red[256];
    const int b = blockIdx.x, tid = threadIdx.x;

    float mx = -3.402823466e38f;
    for (int s = tid; s < S; s += 256)
        if (mask[b * S + s] != 0) mx = fmaxf(mx, scores[(size_t)s * B + b]);
    red[tid] = mx; __syncthreads();
    for (int off = 128; off >= 1; off >>= 1) {
        if (tid < off) red[tid] = fmaxf(red[tid], red[tid + off]);
        __syncthreads();
    }
    mx = red[0]; __syncthreads();

    float sum = 0.f;
    for (int s = tid; s < S; s += 256) {
        float e = 0.f;
        if (mask[b * S + s] != 0) e = __expf(scores[(size_t)s * B + b] - mx);
        attn[(size_t)b * S + s] = e;
        sum += e;
    }
    red[tid] = sum; __syncthreads();
    for (int off = 128; off >= 1; off >>= 1) {
        if (tid < off) red[tid] += red[tid + off];
        __syncthreads();
    }
    const float inv = 1.f / red[0];
    for (int s = tid; s < S; s += 256) attn[(size_t)b * S + s] *= inv;
}

// ---------------------------------------------------------------- context: partials over S chunks
__global__ __launch_bounds__(256) void context_partial(const float* __restrict__ enc,
                                                       const float* __restrict__ attn,
                                                       float* __restrict__ partial) {
    const int b  = blockIdx.x & (B - 1);
    const int hc = (blockIdx.x >> 6) & 3;
    const int sc = blockIdx.x >> 8;              // 0..7
    const int h  = hc * 256 + threadIdx.x;
    const int s0 = sc * (S / 8);
    float acc = 0.f;
    for (int s = s0; s < s0 + S / 8; ++s)
        acc += attn[(size_t)b * S + s] * enc[((size_t)s * B + b) * H + h];
    partial[((size_t)sc * B + b) * H + h] = acc;  // fully owned: no atomics, deterministic
}

__global__ __launch_bounds__(256) void context_reduce(const float* __restrict__ partial,
                                                      float* __restrict__ out) {
    const int i = blockIdx.x * 256 + threadIdx.x;
    float s = 0.f;
    #pragma unroll
    for (int p = 0; p < 8; ++p) s += partial[(size_t)p * B * H + i];
    out[i] = s;
}

// ---------------------------------------------------------------- launcher
extern "C" void kernel_launch(void* const* d_in, const int* in_sizes, int n_in,
                              void* d_out, int out_size, void* d_ws, size_t ws_size,
                              hipStream_t stream) {
    const float* dstate = (const float*)d_in[0];   // (B,H)
    const float* enc    = (const float*)d_in[1];   // (S,B,H)
    const int*   mask   = (const int*)d_in[2];     // (B,S)
    const float* Wp     = (const float*)d_in[3];   // (H,H)
    const float* We     = (const float*)d_in[4];   // (H,H)
    const float* v      = (const float*)d_in[5];   // (H,)

    char* ws = (char*)d_ws;
    unsigned short* we_bf  = (unsigned short*)(ws);                         // 2 MB
    float*          pprev  = (float*)(ws + (size_t)H * H * 2);              // 256 KB
    float*          scores = (float*)(ws + (size_t)H * H * 2 + (size_t)B * H * 4);          // 512 KB
    float*          attn   = (float*)(ws + (size_t)H * H * 2 + (size_t)B * H * 4
                                         + (size_t)B * S * 4);              // 512 KB
    float*          part   = (float*)(ws + (size_t)H * H * 2 + (size_t)B * H * 4
                                         + (size_t)B * S * 8);              // 2 MB

    conv_we_bf16   <<<(H * H) / 1024, 256, 0, stream>>>(We, we_bf);
    proj_prev_kernel<<<(B * H) / 8,   256, 0, stream>>>(dstate, Wp, pprev);
    scores_kernel  <<<(S * B) / 16,   256, 0, stream>>>(enc, we_bf, pprev, v, scores);
    softmax_kernel <<<B,              256, 0, stream>>>(scores, mask, attn);
    context_partial<<<B * 4 * 8,      256, 0, stream>>>(enc, attn, part);
    context_reduce <<<(B * H) / 256,  256, 0, stream>>>(part, (float*)d_out);
}
